// ABINetIterModel_7164005449881
// MI455X (gfx1250) — matrix operand order repaired and checked by reading, not compiled, not run on hardware
//
#include <hip/hip_runtime.h>
#include <hip/hip_bf16.h>
#include <math.h>

// ---------------- problem constants ----------------
#define BB 512
#define TT 32
#define DD 512
#define CC 37
#define HH 8
#define DHH 64
#define FFD 2048
#define BT (BB*TT)          // 16384
#define SCALE_ATT 0.125f    // 1/sqrt(64)

typedef __bf16 bf16;
typedef __bf16 v16bf __attribute__((ext_vector_type(16)));
typedef float  v8f   __attribute__((ext_vector_type(8)));
typedef unsigned int v8u __attribute__((ext_vector_type(8)));

union AFrag { v16bf v; unsigned int u[8]; };
union BFrag { v16bf v; v8u u; };

__device__ __forceinline__ unsigned short f2bf_bits(float f) {
    union { float f; unsigned u; } a; a.f = f;
    unsigned u = a.u;
    unsigned r = (u + 0x7fffu + ((u >> 16) & 1u)) >> 16;   // RNE
    return (unsigned short)r;
}

__device__ __forceinline__ float geluf(float x) {
    float x3 = x * x * x;
    return 0.5f * x * (1.0f + tanhf(0.7978845608028654f * (x + 0.044715f * x3)));
}

// ================= weight repack: f32 [K,N] -> WMMA B-fragment order, bf16 =================
// Packed layout: tile (kc=k/32, nt=n/16): 32 lanes x 8 dwords, lane-contiguous.
// lane l: n = nt*16 + (l&15), khalf g = l>>4; dword j holds bf16 pair (k0 = kc*32+g*16+2j, k0+1).
__global__ void repack_kernel(const float* __restrict__ W, int K, int N, int Kp, int Np,
                              unsigned int* __restrict__ out) {
    int gid = blockIdx.x * 256 + threadIdx.x;
    int total = (Kp >> 5) * (Np >> 4) * 256;
    if (gid >= total) return;
    int j    = gid & 7;
    int lane = (gid >> 3) & 31;
    int tile = gid >> 8;
    int ntiles = Np >> 4;
    int nt = tile % ntiles;
    int kc = tile / ntiles;
    int n  = nt * 16 + (lane & 15);
    int k0 = kc * 32 + (lane >> 4) * 16 + 2 * j;
    unsigned lo = 0, hi = 0;
    if (n < N && k0     < K) lo = f2bf_bits(W[(size_t)k0 * N + n]);
    if (n < N && (k0+1) < K) hi = f2bf_bits(W[(size_t)(k0+1) * N + n]);
    out[gid] = lo | (hi << 16);
}

// ================= generic bf16 WMMA GEMM, 128x64 tile per block (8 waves) =================
// A tile (128x32 bf16) double-buffered in LDS via gfx1250 async global->LDS DMA.
// B fragments double-buffered in registers; k-loop unrolled x2 (ping-pong roles, no movs).
// Cross-wave sync uses a raw split barrier preceded only by s_wait_asynccnt, so register
// prefetches of B are NOT force-waited at the barrier (no loadcnt fence).
// Requires K % 64 == 0 (all GEMMs here: K in {64,512,1024,2048}).
// MODE 0: f32 out (+opt bias)
// MODE 1: f32 out + bf16 mirror (+opt bias)
// MODE 2: bias + gelu -> bf16 only
// MODE 3: bias + sigmoid gate: out_bf = g*aux0 + (1-g)*aux1 (aux row stride = DD)
// MODE 4: bf16 only (+opt bias)
template <int MODE>
__global__ __launch_bounds__(256)
void gemm_bf16_wmma(const bf16* __restrict__ A, const unsigned int* __restrict__ Bpk,
                    int M, int N, int K,
                    const float* __restrict__ bias, int nbias,
                    float* __restrict__ out32, int ld32,
                    bf16* __restrict__ outbf, int ldbf,
                    const float* __restrict__ aux0, const float* __restrict__ aux1) {
    // 2 buffers x 128 rows x 20 dwords (row stride 80B: 16B-aligned chunks, conflict-free reads)
    __shared__ unsigned int Asmem[2][128 * 20];
    const int tid  = threadIdx.x;
    const int wave = tid >> 5;
    const int lane = tid & 31;
    const int m0 = blockIdx.y * 128;
    const int n0 = blockIdx.x * 64;
    const int ntiles = N >> 4;

    v8f acc[4];
    #pragma unroll
    for (int s = 0; s < 4; ++s)
        #pragma unroll
        for (int r = 0; r < 8; ++r) acc[s][r] = 0.0f;

    const int mloc = (wave << 4) + (lane & 15);
    const int g    = lane >> 4;

    // async staging: 128 rows x 64B = 8KB per chunk; 2 lane-b128 chunks per wave
    // cidx = wave*64 + i*32 + lane  (i=0,1) -> row r = cidx>>2, 16B col c = cidx&3
    const int cidx0 = wave * 64 + lane;
    const int cidx1 = cidx0 + 32;
    const int r0 = cidx0 >> 2, c0 = cidx0 & 3;
    const int r1 = cidx1 >> 2, c1 = cidx1 & 3;
    const bf16* gA0 = A + (size_t)(m0 + r0) * K + c0 * 8;
    const bf16* gA1 = A + (size_t)(m0 + r1) * K + c1 * 8;
    const unsigned ldsBase = (unsigned)(unsigned long long)&Asmem[0][0];
    const unsigned lo0 = ldsBase + (unsigned)(r0 * 80 + c0 * 16);
    const unsigned lo1 = ldsBase + (unsigned)(r1 * 80 + c1 * 16);
    const unsigned bufStride = 128 * 20 * 4;

#define ISSUE_A(bufsel, kc)                                                              \
    do {                                                                                 \
        asm volatile("global_load_async_to_lds_b128 %0, %1, off"                         \
                     :: "v"(lo0 + (bufsel) * bufStride),                                 \
                        "v"((unsigned long long)(gA0 + (kc))) : "memory");               \
        asm volatile("global_load_async_to_lds_b128 %0, %1, off"                         \
                     :: "v"(lo1 + (bufsel) * bufStride),                                 \
                        "v"((unsigned long long)(gA1 + (kc))) : "memory");               \
    } while (0)

// own async writes complete, then cross-wave visibility -- WITHOUT a loadcnt fence,
// so in-flight register prefetches of B keep flying across the barrier.
#define ASYNC_BARRIER()                                                                  \
    asm volatile("s_wait_asynccnt 0x0\n\t"                                               \
                 "s_barrier_signal -1\n\t"                                               \
                 "s_barrier_wait -1" ::: "memory")

    // per-subtile packed-B base pointers (chunk 0), advance by kstride dwords per chunk
    const size_t kstride = (size_t)ntiles * 256;
    const unsigned int* Bp[4];
    #pragma unroll
    for (int s = 0; s < 4; ++s)
        Bp[s] = Bpk + ((size_t)((n0 >> 4) + s) * 32 + lane) * 8;

    ISSUE_A(0, 0);

    BFrag b0[4], b1[4];
    #pragma unroll
    for (int s = 0; s < 4; ++s) b0[s].u = *(const v8u*)Bp[s];

    const int nk = K >> 5;           // even for all shapes used here
    for (int t = 0; t < nk; t += 2) {
        // ---- even chunk t: compute from LDS buf0 / regs b0; prefetch t+1 into buf1/b1 ----
        #pragma unroll
        for (int s = 0; s < 4; ++s)
            b1[s].u = *(const v8u*)(Bp[s] + (size_t)(t + 1) * kstride);
        ASYNC_BARRIER();
        ISSUE_A(1, (t + 1) << 5);
        {
            const unsigned* Ab = &Asmem[0][0];
            AFrag af;
            #pragma unroll
            for (int j = 0; j < 8; ++j) {
                int kd = j + g * 8;
                af.u[j] = Ab[mloc * 20 + kd];
            }
            #pragma unroll
            for (int s = 0; s < 4; ++s)
                acc[s] = __builtin_amdgcn_wmma_f32_16x16x32_bf16(
                    false, af.v, false, b0[s].v, (short)0, acc[s], false, false);
        }
        // ---- odd chunk t+1: compute from LDS buf1 / regs b1; prefetch t+2 into buf0/b0 ----
        if (t + 2 < nk) {
            #pragma unroll
            for (int s = 0; s < 4; ++s)
                b0[s].u = *(const v8u*)(Bp[s] + (size_t)(t + 2) * kstride);
        }
        ASYNC_BARRIER();
        if (t + 2 < nk) ISSUE_A(0, (t + 2) << 5);
        {
            const unsigned* Ab = &Asmem[1][0];
            AFrag af;
            #pragma unroll
            for (int j = 0; j < 8; ++j) {
                int kd = j + g * 8;
                af.u[j] = Ab[mloc * 20 + kd];
            }
            #pragma unroll
            for (int s = 0; s < 4; ++s)
                acc[s] = __builtin_amdgcn_wmma_f32_16x16x32_bf16(
                    false, af.v, false, b1[s].v, (short)0, acc[s], false, false);
        }
    }
#undef ISSUE_A
#undef ASYNC_BARRIER

    // epilogue
    #pragma unroll
    for (int s = 0; s < 4; ++s) {
        int col = n0 + s * 16 + (lane & 15);
        float bv = 0.0f;
        if (bias && col < nbias) bv = bias[col];
        #pragma unroll
        for (int r = 0; r < 8; ++r) {
            int row = m0 + (wave << 4) + r + (g << 3);
            float v = acc[s][r] + bv;
            if (MODE == 0) {
                out32[(size_t)row * ld32 + col] = v;
            } else if (MODE == 1) {
                out32[(size_t)row * ld32 + col] = v;
                outbf[(size_t)row * ldbf + col] = (bf16)v;
            } else if (MODE == 2) {
                outbf[(size_t)row * ldbf + col] = (bf16)geluf(v);
            } else if (MODE == 3) {
                float gt  = 1.0f / (1.0f + __expf(-v));
                float vis = aux0[(size_t)row * DD + col];
                float ft  = aux1[(size_t)row * DD + col];
                outbf[(size_t)row * ldbf + col] = (bf16)(gt * vis + (1.0f - gt) * ft);
            } else { // MODE 4
                outbf[(size_t)row * ldbf + col] = (bf16)v;
            }
        }
    }
}

// ================= CTC run-length rearrange =================
__global__ __launch_bounds__(256)
void ctc_kernel(const float* __restrict__ pred, const float* __restrict__ feat, int Wf,
                float* __restrict__ out32, int ld32, int padW,
                bf16* __restrict__ outbf, int ldbf, int bfoff) {
    const int b = blockIdx.x;
    const int tid = threadIdx.x;
    __shared__ int ids[TT];
    __shared__ int slot[TT];

    if (tid < TT) {
        const float* pr = pred + (size_t)(b * TT + tid) * CC;
        int best = 0; float bv = pr[0];
        for (int c = 1; c < CC; ++c) { float v = pr[c]; if (v > bv) { bv = v; best = c; } }
        ids[tid] = best;
    }
    __syncthreads();
    if (tid == 0) {
        int cnt = 0, prev = 0;
        for (int t = 0; t < TT; ++t) {
            int id = ids[t];
            slot[t] = (id != prev && id != 0) ? cnt++ : -1;
            prev = id;
        }
    }
    __syncthreads();
    for (int t = 0; t < TT; ++t) {
        float* orow = out32 + (size_t)(b * TT + t) * ld32;
        for (int c = tid; c < padW; c += 256) orow[c] = 0.0f;
        if (outbf) {
            bf16* brow = outbf + (size_t)(b * TT + t) * ldbf + bfoff;
            for (int c = tid; c < Wf; c += 256) brow[c] = (bf16)0.0f;
        }
    }
    __syncthreads();
    for (int t = 0; t < TT; ++t) {
        int s = slot[t];
        if (s < 0) continue;
        const float* fr = feat + (size_t)(b * TT + t) * Wf;
        float* orow = out32 + (size_t)(b * TT + s) * ld32;
        bf16*  brow = outbf ? outbf + (size_t)(b * TT + s) * ldbf + bfoff : nullptr;
        for (int c = tid; c < Wf; c += 256) {
            float v = fr[c];
            orow[c] = v;
            if (brow) brow[c] = (bf16)v;
        }
    }
}

// ================= softmax over C (ap[.,64] f32 -> p[.,64] bf16, zero-padded) =================
__global__ void softmax_kernel(const float* __restrict__ ap, bf16* __restrict__ p) {
    int row = blockIdx.x * 256 + threadIdx.x;
    if (row >= BT) return;
    const float* a = ap + (size_t)row * 64;
    float m = a[0];
    for (int c = 1; c < CC; ++c) m = fmaxf(m, a[c]);
    float e[CC]; float s = 0.0f;
    for (int c = 0; c < CC; ++c) { e[c] = __expf(a[c] - m); s += e[c]; }
    float inv = 1.0f / s;
    bf16* pr = p + (size_t)row * 64;
    for (int c = 0; c < CC; ++c) pr[c] = (bf16)(e[c] * inv);
    for (int c = CC; c < 64; ++c) pr[c] = (bf16)0.0f;
}

// ================= LayerNorm(a + b), wave-per-row =================
__global__ __launch_bounds__(256)
void ln_kernel(const float* __restrict__ A, const float* __restrict__ Bv,
               const float* __restrict__ gam, const float* __restrict__ bet,
               float* __restrict__ out32, bf16* __restrict__ outbf) {
    int wave = threadIdx.x >> 5, lane = threadIdx.x & 31;
    int row = blockIdx.x * 8 + wave;
    const float* ar = A  + (size_t)row * DD;
    const float* br = Bv + (size_t)row * DD;
    float xv[16]; float s = 0.0f, ss = 0.0f;
    #pragma unroll
    for (int i = 0; i < 16; ++i) {
        int c = lane + i * 32;
        float v = ar[c] + br[c];
        xv[i] = v; s += v; ss += v * v;
    }
    #pragma unroll
    for (int o = 16; o > 0; o >>= 1) { s += __shfl_xor(s, o, 32); ss += __shfl_xor(ss, o, 32); }
    float mean = s * (1.0f / DD);
    float var  = ss * (1.0f / DD) - mean * mean;
    float rstd = rsqrtf(var + 1e-5f);
    #pragma unroll
    for (int i = 0; i < 16; ++i) {
        int c = lane + i * 32;
        float y = (xv[i] - mean) * rstd * gam[c] + bet[c];
        if (out32) out32[(size_t)row * DD + c] = y;
        if (outbf) outbf[(size_t)row * DD + c] = (bf16)y;
    }
}

// ================= attention: per (b,h): softmax(QK^T*s)@V, LDS-staged Q/K/V =================
__global__ __launch_bounds__(256)
void attention_kernel(const bf16* __restrict__ qkv, bf16* __restrict__ obf) {
    int bh = blockIdx.x;
    int b = bh / HH, h = bh % HH;
    const int tid = threadIdx.x;
    __shared__ float Qs[TT][DHH + 1];
    __shared__ float Ks[TT][DHH + 1];
    __shared__ float Vs[TT][DHH + 1];
    __shared__ float S[TT][TT + 1];

    const size_t rowbase = (size_t)b * TT;
    const int qoff = h * DHH, koff = DD + h * DHH, voff = 2 * DD + h * DHH;

    #pragma unroll
    for (int i = 0; i < 8; ++i) {
        int idx = tid + i * 256;          // 0..2047
        int t = idx >> 6, d = idx & 63;
        const bf16* base = qkv + (rowbase + t) * (3 * DD);
        Qs[t][d] = (float)base[qoff + d];
        Ks[t][d] = (float)base[koff + d];
        Vs[t][d] = (float)base[voff + d];
    }
    __syncthreads();
    #pragma unroll
    for (int i = 0; i < 4; ++i) {
        int idx = tid + i * 256;          // 0..1023
        int q = idx >> 5, k = idx & 31;
        float sum = 0.0f;
        #pragma unroll
        for (int d = 0; d < DHH; ++d) sum += Qs[q][d] * Ks[k][d];
        S[q][k] = sum * SCALE_ATT;
    }
    __syncthreads();
    if (tid < TT) {
        float m = S[tid][0];
        for (int k = 1; k < TT; ++k) m = fmaxf(m, S[tid][k]);
        float sum = 0.0f;
        for (int k = 0; k < TT; ++k) { float e = __expf(S[tid][k] - m); S[tid][k] = e; sum += e; }
        float inv = 1.0f / sum;
        for (int k = 0; k < TT; ++k) S[tid][k] *= inv;
    }
    __syncthreads();
    #pragma unroll
    for (int i = 0; i < 8; ++i) {
        int idx = tid + i * 256;          // 0..2047
        int q = idx >> 6, d = idx & 63;
        float sum = 0.0f;
        #pragma unroll
        for (int k = 0; k < TT; ++k) sum += S[q][k] * Vs[k][d];
        obf[(rowbase + q) * DD + h * DHH + d] = (bf16)sum;
    }
}

// ================= final copy ap[.,64] -> d_out [B,T,C] =================
__global__ void out_copy_kernel(const float* __restrict__ ap, float* __restrict__ out) {
    int i = blockIdx.x * 256 + threadIdx.x;
    if (i >= BT * CC) return;
    int row = i / CC, c = i % CC;
    out[i] = ap[(size_t)row * 64 + c];
}

// ======================================================================================
extern "C" void kernel_launch(void* const* d_in, const int* in_sizes, int n_in,
                              void* d_out, int out_size, void* d_ws, size_t ws_size,
                              hipStream_t stream) {
    const float* vision_feature = (const float*)d_in[0];
    const float* vision_pred    = (const float*)d_in[1];
    const float* W_e   = (const float*)d_in[2];
    const float* W_qkv = (const float*)d_in[3];
    const float* W_o   = (const float*)d_in[4];
    const float* ln1g  = (const float*)d_in[5];
    const float* ln1b  = (const float*)d_in[6];
    const float* W1    = (const float*)d_in[7];
    const float* b1    = (const float*)d_in[8];
    const float* W2    = (const float*)d_in[9];
    const float* b2    = (const float*)d_in[10];
    const float* ln2g  = (const float*)d_in[11];
    const float* ln2b  = (const float*)d_in[12];
    const float* Wp    = (const float*)d_in[13];
    const float* bp    = (const float*)d_in[14];
    const float* Watt  = (const float*)d_in[15];
    const float* batt  = (const float*)d_in[16];
    const float* Wac   = (const float*)d_in[17];
    const float* bac   = (const float*)d_in[18];

    char* ws = (char*)d_ws;
    size_t off = 0;
    auto alloc = [&](size_t bytes) -> void* {
        void* p = ws + off;
        off += (bytes + 255) & ~(size_t)255;
        return p;
    };

    // packed weights (bf16, dword-packed)
    unsigned* We_pk   = (unsigned*)alloc((size_t)64   * 512  * 2);
    unsigned* Wqkv_pk = (unsigned*)alloc((size_t)512  * 1536 * 2);
    unsigned* Wo_pk   = (unsigned*)alloc((size_t)512  * 512  * 2);
    unsigned* W1_pk   = (unsigned*)alloc((size_t)512  * 2048 * 2);
    unsigned* W2_pk   = (unsigned*)alloc((size_t)2048 * 512  * 2);
    unsigned* Wp_pk   = (unsigned*)alloc((size_t)512  * 512  * 2);
    unsigned* Watt_pk = (unsigned*)alloc((size_t)1024 * 512  * 2);
    unsigned* Wac_pk  = (unsigned*)alloc((size_t)512  * 64   * 2);

    // persistent activations
    float* vis_f32 = (float*)alloc((size_t)BT * DD * 4);
    bf16*  fcat    = (bf16*) alloc((size_t)BT * 1024 * 2);   // [feat_bf | vis_bf]
    float* ap      = (float*)alloc((size_t)BT * 64 * 4);
    bf16*  p_bf    = (bf16*) alloc((size_t)BT * 64 * 2);

    // reusable arenas
    bf16*  S1 = (bf16*) alloc((size_t)BT * FFD * 2);         // qkv_bf / h_bf
    float* S2 = (float*)alloc((size_t)BT * DD * 4);          // x_f32 -> feat_f32
    bf16*  S3 = (bf16*) alloc((size_t)BT * DD * 2);          // x_bf/o_bf/x1_bf/x2_bf/fused_bf
    float* S4 = (float*)alloc((size_t)BT * DD * 4);          // attn_f32 -> ff_f32
    float* S5 = (float*)alloc((size_t)BT * DD * 4);          // x1_f32

    (void)in_sizes; (void)n_in; (void)out_size; (void)ws_size;

    auto rp = [&](const float* W, int K, int N, int Kp, int Np, unsigned* outp) {
        int total = (Kp / 32) * (Np / 16) * 256;
        repack_kernel<<<(total + 255) / 256, 256, 0, stream>>>(W, K, N, Kp, Np, outp);
    };
    rp(W_e,   37,   512,  64,   512,  We_pk);
    rp(W_qkv, 512,  1536, 512,  1536, Wqkv_pk);
    rp(W_o,   512,  512,  512,  512,  Wo_pk);
    rp(W1,    512,  2048, 512,  2048, W1_pk);
    rp(W2,    2048, 512,  2048, 512,  W2_pk);
    rp(Wp,    512,  512,  512,  512,  Wp_pk);
    rp(Watt,  1024, 512,  1024, 512,  Watt_pk);
    rp(Wac,   512,  37,   512,  64,   Wac_pk);

    ctc_kernel<<<BB, 256, 0, stream>>>(vision_pred, vision_feature, DD,
                                       vis_f32, DD, DD, fcat, 1024, DD);
    ctc_kernel<<<BB, 256, 0, stream>>>(vision_pred, vision_pred, CC,
                                       ap, 64, 64, nullptr, 0, 0);

    dim3 blk(256);
    for (int it = 0; it < 3; ++it) {
        softmax_kernel<<<BT / 256, blk, 0, stream>>>(ap, p_bf);

        // x = p @ W_e               -> f32(S2) + bf16(S3)
        gemm_bf16_wmma<1><<<dim3(512 / 64, BT / 128), blk, 0, stream>>>(
            p_bf, We_pk, BT, 512, 64, nullptr, 0, S2, 512, S3, 512, nullptr, nullptr);

        // qkv = x @ W_qkv           -> bf16(S1)
        gemm_bf16_wmma<4><<<dim3(1536 / 64, BT / 128), blk, 0, stream>>>(
            S3, Wqkv_pk, BT, 1536, 512, nullptr, 0, nullptr, 0, S1, 1536, nullptr, nullptr);

        // attention -> o_bf (S3)
        attention_kernel<<<BB * HH, blk, 0, stream>>>(S1, S3);

        // attn = o @ W_o            -> f32(S4)
        gemm_bf16_wmma<0><<<dim3(512 / 64, BT / 128), blk, 0, stream>>>(
            S3, Wo_pk, BT, 512, 512, nullptr, 0, S4, 512, nullptr, 0, nullptr, nullptr);

        // x1 = LN(x + attn)         -> f32(S5) + bf16(S3)
        ln_kernel<<<BT / 8, blk, 0, stream>>>(S2, S4, ln1g, ln1b, S5, S3);

        // h = gelu(x1 @ W1 + b1)    -> bf16(S1)
        gemm_bf16_wmma<2><<<dim3(2048 / 64, BT / 128), blk, 0, stream>>>(
            S3, W1_pk, BT, 2048, 512, b1, 2048, nullptr, 0, S1, 2048, nullptr, nullptr);

        // ff = h @ W2 + b2          -> f32(S4)
        gemm_bf16_wmma<0><<<dim3(512 / 64, BT / 128), blk, 0, stream>>>(
            S1, W2_pk, BT, 512, 2048, b2, 512, S4, 512, nullptr, 0, nullptr, nullptr);

        // x2 = LN(x1 + ff)          -> bf16(S3)
        ln_kernel<<<BT / 8, blk, 0, stream>>>(S5, S4, ln2g, ln2b, nullptr, S3);

        // feat = x2 @ Wp + bp       -> f32(S2) + bf16 into fcat[:, :512]
        gemm_bf16_wmma<1><<<dim3(512 / 64, BT / 128), blk, 0, stream>>>(
            S3, Wp_pk, BT, 512, 512, bp, 512, S2, 512, fcat, 1024, nullptr, nullptr);

        // g = sigmoid(fcat @ Watt + batt); fused = g*vis + (1-g)*feat -> bf16(S3)
        gemm_bf16_wmma<3><<<dim3(512 / 64, BT / 128), blk, 0, stream>>>(
            fcat, Watt_pk, BT, 512, 1024, batt, 512, nullptr, 0, S3, 512, vis_f32, S2);

        // ap = fused @ Wac + bac    -> f32(ap, padded 64)
        gemm_bf16_wmma<0><<<dim3(64 / 64, BT / 128), blk, 0, stream>>>(
            S3, Wac_pk, BT, 64, 512, bac, CC, ap, 64, nullptr, 0, nullptr, nullptr);
    }

    out_copy_kernel<<<(BT * CC + 255) / 256, blk, 0, stream>>>(ap, (float*)d_out);
}
